// SingleHeadAttention_55705725829414
// MI455X (gfx1250) — matrix operand order repaired and checked by reading, not compiled, not run on hardware
//
#include <hip/hip_runtime.h>
#include <hip/hip_bf16.h>

typedef __attribute__((ext_vector_type(16))) _Float16 v16h;
typedef __attribute__((ext_vector_type(8)))  _Float16 v8h;
typedef __attribute__((ext_vector_type(4)))  _Float16 v4h;
typedef __attribute__((ext_vector_type(8)))  float    v8f;

#define EMBED 1024
#define ADIM  1024
#define BATCH 4
#define SEQ   4096
#define NROWS (BATCH * SEQ)

__device__ __forceinline__ v16h cat16(v8h lo, v8h hi) {
  v16h r;
#pragma unroll
  for (int t = 0; t < 8; ++t) { r[t] = lo[t]; r[8 + t] = hi[t]; }
  return r;
}

// ---------------------------------------------------------------------------
// Stage 0: fp32 -> fp16 elementwise convert (vectorized 4-wide)
// ---------------------------------------------------------------------------
__global__ void cvt_f32_f16(const float* __restrict__ src,
                            _Float16* __restrict__ dst, int n) {
  int i = (blockIdx.x * blockDim.x + threadIdx.x) * 4;
  if (i < n) {
    float4 v = *(const float4*)(src + i);
    v4h h;
    h[0] = (_Float16)v.x; h[1] = (_Float16)v.y;
    h[2] = (_Float16)v.z; h[3] = (_Float16)v.w;
    *(v4h*)(dst + i) = h;
  }
}

// ---------------------------------------------------------------------------
// Stage 1: projection GEMM  Out[n,a] = sum_e X[n,e] * W[a,e]
//   TRANSPOSED=0: Out row-major [NROWS, ADIM]   (Q, K)
//   TRANSPOSED=1: Out per-batch transposed [BATCH][ADIM][SEQ]  (V^T)
// Wave tile: M=16, N=64 (4 wmma accumulators), K-loop over EMBED (rolled to
// keep register pressure low -> no spills; operands stream from L0/L2).
// ---------------------------------------------------------------------------
template <int TRANSPOSED>
__global__ void proj_kernel(const _Float16* __restrict__ X,
                            const _Float16* __restrict__ W,
                            _Float16* __restrict__ Out) {
  const int lane = threadIdx.x & 31;
  const int wave = threadIdx.x >> 5;
  const int lm   = lane & 15;
  const int hf   = lane >> 4;                 // half-wave: 0 or 1
  const int m0   = (blockIdx.y * 4 + wave) * 16;
  const int n0   = blockIdx.x * 64;

  v8f acc[4] = {};
  const _Float16* xrow = X + (size_t)(m0 + lm) * EMBED;
#pragma unroll 1
  for (int ks = 0; ks < EMBED; ks += 32) {
    v16h a = *(const v16h*)(xrow + ks + hf * 16);
#pragma unroll
    for (int j = 0; j < 4; ++j) {
      v16h b = *(const v16h*)(W + (size_t)(n0 + j * 16 + lm) * EMBED + ks + hf * 16);
      acc[j] = __builtin_amdgcn_wmma_f32_16x16x32_f16(false, a, false, b,
                                                      (short)0, acc[j], false, false);
    }
  }
  if (TRANSPOSED == 0) {
#pragma unroll
    for (int j = 0; j < 4; ++j) {
      int col = n0 + j * 16 + lm;
#pragma unroll
      for (int i = 0; i < 8; ++i) {
        int row = m0 + i + hf * 8;
        Out[(size_t)row * ADIM + col] = (_Float16)acc[j][i];
      }
    }
  } else {
    int bidx = m0 / SEQ;
    int s0   = (m0 % SEQ) + hf * 8;
#pragma unroll
    for (int j = 0; j < 4; ++j) {
      int col = n0 + j * 16 + lm;
      v8h o;
#pragma unroll
      for (int i = 0; i < 8; ++i) o[i] = (_Float16)acc[j][i];
      *(v8h*)(Out + ((size_t)bidx * ADIM + col) * SEQ + s0) = o;
    }
  }
}

// ---------------------------------------------------------------------------
// Stage 2: causal scores + softmax statistics.
// One wave owns a 16-row query tile. Outer loop: 64 k-columns per step
// (4 accumulators share one A-operand load -> 2x arithmetic intensity).
// Inner contraction loop is rolled (#pragma unroll 1) so the Q-tile loads
// stay inside the loop and are NOT hoisted/spilled to scratch.
// Scores spilled as f16 to Ph; then exp/sum sweep rewrites Ph in place.
// ---------------------------------------------------------------------------
__global__ void scores_softmax(const _Float16* __restrict__ Qh,
                               const _Float16* __restrict__ Kh,
                               _Float16* __restrict__ Ph,
                               float* __restrict__ Linv) {
  __shared__ float msh[4][16];
  const int lane = threadIdx.x & 31;
  const int wave = threadIdx.x >> 5;
  const int lm   = lane & 15;
  const int hf   = lane >> 4;
  const int b    = blockIdx.y;
  const int q0   = (blockIdx.x * 4 + wave) * 16;
  const int kend = q0 + 16;                 // causal limit (exclusive)
  const float scale = 0.03125f;             // 1/sqrt(1024)

  const _Float16* Qrow  = Qh + (size_t)(b * SEQ + q0 + lm) * ADIM;
  _Float16*       Ptile = Ph + ((size_t)b * SEQ + q0) * (size_t)SEQ;

  float vmax[8];
#pragma unroll
  for (int i = 0; i < 8; ++i) vmax[i] = -INFINITY;

#pragma unroll 1
  for (int kb = 0; kb < kend; kb += 64) {
    v8f acc[4] = {};
    const _Float16* krb = Kh + (size_t)(b * SEQ + kb + lm) * ADIM + hf * 16;
    if (kb + 64 < kend) {                          // prefetch next K block
      __builtin_prefetch(krb + (size_t)64 * ADIM, 0, 1);
    }
#pragma unroll 1
    for (int ac = 0; ac < ADIM; ac += 32) {
      v16h a  = *(const v16h*)(Qrow + ac + hf * 16);
#pragma unroll
      for (int t = 0; t < 4; ++t) {
        v16h bt = *(const v16h*)(krb + (size_t)(t * 16) * ADIM + ac);
        acc[t] = __builtin_amdgcn_wmma_f32_16x16x32_f16(false, a, false, bt,
                                                        (short)0, acc[t], false, false);
      }
    }
#pragma unroll
    for (int t = 0; t < 4; ++t) {
      int n = kb + t * 16 + lm;
#pragma unroll
      for (int i = 0; i < 8; ++i) {
        int qg = q0 + i + hf * 8;
        float s = (n <= qg) ? acc[t][i] * scale : -INFINITY;
        vmax[i] = fmaxf(vmax[i], s);
        Ptile[(size_t)(i + hf * 8) * SEQ + n] = (_Float16)s;
      }
    }
  }

  // Row max: reduce across the 16 lanes of each half-wave.
#pragma unroll
  for (int i = 0; i < 8; ++i) {
    float m = vmax[i];
    m = fmaxf(m, __shfl_xor(m, 1, 32));
    m = fmaxf(m, __shfl_xor(m, 2, 32));
    m = fmaxf(m, __shfl_xor(m, 4, 32));
    m = fmaxf(m, __shfl_xor(m, 8, 32));
    vmax[i] = m;
  }
  if (lm == 0) {
#pragma unroll
    for (int i = 0; i < 8; ++i) msh[wave][hf * 8 + i] = vmax[i];
  }
  __threadfence_block();   // make spilled scores + msh visible wave-wide
  __syncthreads();

  // Sweep 2: lane L handles row (L&15); halves split columns in 16-chunks.
  const int r = lm;
  const float m = msh[wave][r];
  _Float16* prow = Ptile + (size_t)r * SEQ;
  const int kcols = (kend + 63) & ~63;      // region written above
  float sum = 0.f;
#pragma unroll 1
  for (int c = hf * 16; c < kcols; c += 32) {
    v8h p0 = *(const v8h*)(prow + c);
    v8h p1 = *(const v8h*)(prow + c + 8);
    v8h e0, e1;
#pragma unroll
    for (int t = 0; t < 8; ++t) {
      float x0 = __expf((float)p0[t] - m);
      float x1 = __expf((float)p1[t] - m);
      sum += x0 + x1;
      e0[t] = (_Float16)x0;
      e1[t] = (_Float16)x1;
    }
    *(v8h*)(prow + c)     = e0;
    *(v8h*)(prow + c + 8) = e1;
  }
  sum += __shfl_xor(sum, 16, 32);
  if (hf == 0) Linv[b * SEQ + q0 + r] = 1.0f / sum;
}

// ---------------------------------------------------------------------------
// Stage 3: Out = (P @ V) * (1/l), rounded to 4 decimals, fp32.
// Wave tile M=16 (q) x N=64 (a); causal k-loop (rolled). B-operand from V^T.
// ---------------------------------------------------------------------------
__global__ void pv_out(const _Float16* __restrict__ Ph,
                       const _Float16* __restrict__ Vth,
                       const float* __restrict__ Linv,
                       float* __restrict__ Out) {
  const int lane = threadIdx.x & 31;
  const int wave = threadIdx.x >> 5;
  const int lm   = lane & 15;
  const int hf   = lane >> 4;
  const int b    = blockIdx.z;
  const int q0   = (blockIdx.y * 4 + wave) * 16;
  const int n0   = blockIdx.x * 64;
  const int kcols = ((q0 + 16) + 31) & ~31;   // subset of region scores wrote

  const _Float16* Prow = Ph + ((size_t)b * SEQ + q0 + lm) * (size_t)SEQ;
  v8f acc[4] = {};
#pragma unroll 1
  for (int kb = 0; kb < kcols; kb += 32) {
    v16h a  = *(const v16h*)(Prow + kb + hf * 16);
#pragma unroll
    for (int j = 0; j < 4; ++j) {
      v16h bb = *(const v16h*)(Vth + ((size_t)b * ADIM + n0 + j * 16 + lm) * SEQ
                               + kb + hf * 16);
      acc[j] = __builtin_amdgcn_wmma_f32_16x16x32_f16(false, a, false, bb,
                                                      (short)0, acc[j], false, false);
    }
  }
  float inv[8];
#pragma unroll
  for (int i = 0; i < 8; ++i) inv[i] = Linv[b * SEQ + q0 + i + hf * 8];
#pragma unroll
  for (int j = 0; j < 4; ++j) {
    int col = n0 + j * 16 + lm;
#pragma unroll
    for (int i = 0; i < 8; ++i) {
      int row = q0 + i + hf * 8;
      float v = acc[j][i] * inv[i];
      v = __builtin_rintf(v * 1.0e4f) * 1.0e-4f;   // round to 4 decimals
      Out[((size_t)b * SEQ + row) * ADIM + col] = v;
    }
  }
}

// ---------------------------------------------------------------------------
extern "C" void kernel_launch(void* const* d_in, const int* in_sizes, int n_in,
                              void* d_out, int out_size, void* d_ws, size_t ws_size,
                              hipStream_t stream) {
  (void)in_sizes; (void)n_in; (void)out_size; (void)ws_size;
  const float* X  = (const float*)d_in[0];   // embedded [B,S,E]
  const float* Wk = (const float*)d_in[1];   // [A,E]
  const float* Wq = (const float*)d_in[2];   // [A,E]
  const float* Wv = (const float*)d_in[3];   // [A,E]
  float* Out = (float*)d_out;

  char* ws = (char*)d_ws;
  size_t off = 0;
  _Float16* Xh  = (_Float16*)(ws + off); off += (size_t)NROWS * EMBED * 2;       // 32 MB
  _Float16* Wkh = (_Float16*)(ws + off); off += (size_t)ADIM * EMBED * 2;        //  2 MB
  _Float16* Wqh = (_Float16*)(ws + off); off += (size_t)ADIM * EMBED * 2;
  _Float16* Wvh = (_Float16*)(ws + off); off += (size_t)ADIM * EMBED * 2;
  _Float16* Qh  = (_Float16*)(ws + off); off += (size_t)NROWS * ADIM * 2;        // 32 MB
  _Float16* Kh  = (_Float16*)(ws + off); off += (size_t)NROWS * ADIM * 2;        // 32 MB
  _Float16* Vth = (_Float16*)(ws + off); off += (size_t)BATCH * ADIM * SEQ * 2;  // 32 MB
  _Float16* Phb = (_Float16*)(ws + off); off += (size_t)BATCH * SEQ * SEQ * 2;   // 128 MB
  float*    Linv= (float*)(ws + off);    off += (size_t)NROWS * sizeof(float);

  // Stage 0: convert to f16
  cvt_f32_f16<<<dim3((NROWS * EMBED) / 1024), 256, 0, stream>>>(X, Xh, NROWS * EMBED);
  cvt_f32_f16<<<dim3((ADIM * EMBED) / 1024), 256, 0, stream>>>(Wk, Wkh, ADIM * EMBED);
  cvt_f32_f16<<<dim3((ADIM * EMBED) / 1024), 256, 0, stream>>>(Wq, Wqh, ADIM * EMBED);
  cvt_f32_f16<<<dim3((ADIM * EMBED) / 1024), 256, 0, stream>>>(Wv, Wvh, ADIM * EMBED);

  // Stage 1: projections (Q,K row-major; V transposed)
  dim3 pgrid(ADIM / 64, NROWS / 64);
  proj_kernel<0><<<pgrid, 128, 0, stream>>>(Xh, Wqh, Qh);
  proj_kernel<0><<<pgrid, 128, 0, stream>>>(Xh, Wkh, Kh);
  proj_kernel<1><<<pgrid, 128, 0, stream>>>(Xh, Wvh, Vth);

  // Stage 2: causal scores + softmax
  scores_softmax<<<dim3(SEQ / 64, BATCH), 128, 0, stream>>>(Qh, Kh, Phb, Linv);

  // Stage 3: P @ V, normalize, round, store fp32
  pv_out<<<dim3(ADIM / 64, SEQ / 64, BATCH), 128, 0, stream>>>(Phb, Vth, Linv, Out);
}
